// SGFormer_63513976373670
// MI455X (gfx1250) — matrix operand order repaired and checked by reading, not compiled, not run on hardware
//
#include <hip/hip_runtime.h>
#include <hip/hip_bf16.h>

// ---------------------------------------------------------------------------
// SGFormer (transformer branch + GCN branch) for MI455X / gfx1250.
// All dense matmuls run on V_WMMA_F32_16X16X32_F16 (f32 accum), with f32
// activations converted to f16 on the fly into LDS. kv = k^T v reduction also
// uses WMMA over LDS-transposed chunks + f32 global atomics.
// ---------------------------------------------------------------------------

typedef _Float16 h8  __attribute__((ext_vector_type(8)));
typedef _Float16 h16 __attribute__((ext_vector_type(16)));
typedef float    f8  __attribute__((ext_vector_type(8)));
typedef float    f4  __attribute__((ext_vector_type(4)));

#define N_NODES 50000
#define N_EDGES 500000
#define DD      128
#define NPAD    50048   // multiple of 64 for the kv chunk loop

static constexpr float kBN = 0.9999950000374997f;  // 1/sqrt(1 + 1e-5)

__device__ __forceinline__ h16 cat8(h8 lo, h8 hi) {
  return __builtin_shufflevector(lo, hi, 0,1,2,3,4,5,6,7,8,9,10,11,12,13,14,15);
}

__device__ __forceinline__ float waveRedSum(float v) {
#pragma unroll
  for (int m = 16; m >= 1; m >>= 1) v += __shfl_xor(v, m, 32);
  return v;
}

// ---------------------------------------------------------------- utilities
__global__ void fill_f32(float* __restrict__ p, int n, float v) {
  int i = blockIdx.x * 256 + threadIdx.x;
  if (i < n) p[i] = v;
}

__global__ void copy_f32(const float* __restrict__ s, float* __restrict__ d, int n) {
  int i = blockIdx.x * 256 + threadIdx.x;
  if (i < n) d[i] = s[i];
}

// dst[n*128+k] = (f16) src[k*128+n]   (transpose + convert, one 128x128 matrix)
__global__ void wprep(const float* __restrict__ src, _Float16* __restrict__ dst) {
  int i = blockIdx.x * 256 + threadIdx.x;          // 16384 elements
  int k = i >> 7, n = i & 127;
  dst[n * DD + k] = (_Float16)src[k * DD + n];
}

// --------------------------------------------------------------- main GEMM
// out[N x 128] = A[N x 128] @ W (Wt is W^T in f16, Wt[n][k])
// flags: 1 = +bias[col]   2 = BatchNorm(eval)+ReLU   4 = += addScale*addP
__global__ __launch_bounds__(256) void gemm_nt128(
    const float* __restrict__ A, const _Float16* __restrict__ Wt,
    const float* __restrict__ bias, const float* __restrict__ bnG,
    const float* __restrict__ bnB, const float* __restrict__ addP,
    float addScale, float* __restrict__ out, int flags)
{
  __shared__ _Float16 As[16][136];                 // +8 pad: no bank conflicts
  const int t    = threadIdx.x;
  const int row0 = blockIdx.x * 16;

  { // stage 16x128 f32 -> f16 LDS tile
    int r = t >> 4, c0 = (t & 15) * 8;
    const float* src = A + (size_t)(row0 + r) * DD + c0;
    f4 v0 = *(const f4*)src;
    f4 v1 = *(const f4*)(src + 4);
    h8 h;
    h[0]=(_Float16)v0[0]; h[1]=(_Float16)v0[1]; h[2]=(_Float16)v0[2]; h[3]=(_Float16)v0[3];
    h[4]=(_Float16)v1[0]; h[5]=(_Float16)v1[1]; h[6]=(_Float16)v1[2]; h[7]=(_Float16)v1[3];
    *(h8*)&As[r][c0] = h;
  }
  __syncthreads();

  const int wave  = t >> 5, lane = t & 31;
  const int col0  = wave * 16;
  const int mA    = lane & 15;
  const int kselA = (lane >> 4) * 8;               // A frag K base within 32
  const int nB    = col0 + (lane & 15);
  const int kselB = (lane >> 4) * 8;               // B frag K base within 32

  f8 acc = {0.f,0.f,0.f,0.f,0.f,0.f,0.f,0.f};
#pragma unroll
  for (int kk = 0; kk < 4; ++kk) {
    int ka = kk * 32 + kselA;
    h16 a = cat8(*(const h8*)&As[mA][ka], *(const h8*)&As[mA][ka + 16]);
    const _Float16* bp = Wt + (size_t)nB * DD + kk * 32 + kselB;
    h16 b = cat8(*(const h8*)bp, *(const h8*)(bp + 16));
    acc = __builtin_amdgcn_wmma_f32_16x16x32_f16(false, a, false, b,
                                                 (short)0, acc, false, false);
  }

  // epilogue: C layout — VGPR r: lanes0-15 M=r, lanes16-31 M=r+8, N=lane&15
  const int colg  = col0 + (lane & 15);
  const int rbase = row0 + ((lane >> 4) * 8);
  float bv = (flags & 1) ? bias[colg] : 0.f;
  float sc = 1.f, sh = 0.f;
  if (flags & 2) { sc = bnG[colg] * kBN; sh = bnB[colg]; }
#pragma unroll
  for (int r = 0; r < 8; ++r) {
    int rowg = rbase + r;
    float v = acc[r] + bv;
    if (flags & 4) v += addScale * addP[(size_t)rowg * DD + colg];
    if (flags & 2) { v = fmaxf(v * sc + sh, 0.f); }
    out[(size_t)rowg * DD + colg] = v;
  }
}

// ------------------------------------------------- kv[m][d] = sum_n k[n][m]*v[n][d]
__global__ __launch_bounds__(256) void kv_accum(
    const float* __restrict__ K, const float* __restrict__ V,
    float* __restrict__ kv, int nChunks)
{
  __shared__ _Float16 Kt[128][72];   // Kt[m][n]  (transposed chunk, f16)
  __shared__ _Float16 Vt[128][72];   // Vt[d][n]
  const int t = threadIdx.x, wave = t >> 5, lane = t & 31;

  f8 acc[8];
#pragma unroll
  for (int c = 0; c < 8; ++c)
    acc[c] = (f8){0.f,0.f,0.f,0.f,0.f,0.f,0.f,0.f};

  for (int ch = blockIdx.x; ch < nChunks; ch += gridDim.x) {
    __syncthreads();
    { // stage 64-node chunk transposed
      int n = t >> 2, d0 = (t & 3) * 32;
      const float* kp = K + (size_t)(ch * 64 + n) * DD + d0;
      const float* vp = V + (size_t)(ch * 64 + n) * DD + d0;
#pragma unroll
      for (int j = 0; j < 32; j += 4) {
        f4 kq = *(const f4*)(kp + j);
        f4 vq = *(const f4*)(vp + j);
#pragma unroll
        for (int u = 0; u < 4; ++u) {
          Kt[d0 + j + u][n] = (_Float16)kq[u];
          Vt[d0 + j + u][n] = (_Float16)vq[u];
        }
      }
    }
    __syncthreads();

    const int mA    = wave * 16 + (lane & 15);
    const int kselA = (lane >> 4) * 8;
    const int nsel  = lane & 15;
    const int kselB = (lane >> 4) * 8;
#pragma unroll
    for (int kk = 0; kk < 2; ++kk) {             // K = 64 per chunk
      int ka = kk * 32 + kselA;
      h16 a = cat8(*(const h8*)&Kt[mA][ka], *(const h8*)&Kt[mA][ka + 16]);
#pragma unroll
      for (int ct = 0; ct < 8; ++ct) {
        int nb = ct * 16 + nsel;
        int kb = kk * 32 + kselB;
        h16 b = cat8(*(const h8*)&Vt[nb][kb], *(const h8*)&Vt[nb][kb + 16]);
        acc[ct] = __builtin_amdgcn_wmma_f32_16x16x32_f16(false, a, false, b,
                                                         (short)0, acc[ct], false, false);
      }
    }
  }
#pragma unroll
  for (int ct = 0; ct < 8; ++ct)
#pragma unroll
    for (int r = 0; r < 8; ++r) {
      int m = wave * 16 + ((lane >> 4) * 8) + r;
      int d = ct * 16 + (lane & 15);
      atomicAdd(&kv[m * DD + d], acc[ct][r]);
    }
}

// ksum[m] = sum_n K[n][m]
__global__ __launch_bounds__(256) void ksum128(const float* __restrict__ K,
                                               float* __restrict__ ksum, int n) {
  __shared__ float red[256];
  int m = blockIdx.x;
  float s = 0.f;
  for (int i = threadIdx.x; i < n; i += 256) s += K[(size_t)i * DD + m];
  red[threadIdx.x] = s;
  __syncthreads();
  for (int o = 128; o > 0; o >>= 1) {
    if (threadIdx.x < o) red[threadIdx.x] += red[threadIdx.x + o];
    __syncthreads();
  }
  if (threadIdx.x == 0) ksum[m] = red[0];
}

// ----------------------------------------------------------- rowwise kernels
// y = relu(LN(x)*g + b), in place; one wave per 128-wide row
__global__ __launch_bounds__(256) void ln_relu_rows(float* __restrict__ X,
    const float* __restrict__ g, const float* __restrict__ b, int nrows) {
  int row = blockIdx.x * 8 + (threadIdx.x >> 5);
  if (row >= nrows) return;
  int lane = threadIdx.x & 31, c = lane * 4;
  f4 v = *(f4*)(X + (size_t)row * DD + c);
  float mu = waveRedSum(v[0] + v[1] + v[2] + v[3]) * (1.f / 128.f);
  float d0 = v[0]-mu, d1 = v[1]-mu, d2 = v[2]-mu, d3 = v[3]-mu;
  float var = waveRedSum(d0*d0 + d1*d1 + d2*d2 + d3*d3) * (1.f / 128.f);
  float rs = rsqrtf(var + 1e-5f);
  f4 o;
  o[0] = fmaxf(d0*rs*g[c+0] + b[c+0], 0.f);
  o[1] = fmaxf(d1*rs*g[c+1] + b[c+1], 0.f);
  o[2] = fmaxf(d2*rs*g[c+2] + b[c+2], 0.f);
  o[3] = fmaxf(d3*rs*g[c+3] + b[c+3], 0.f);
  *(f4*)(X + (size_t)row * DD + c) = o;
}

// x = (x==0 ? 1e-6 : x) / ||x||, in place
__global__ __launch_bounds__(256) void qknorm_rows(float* __restrict__ X, int nrows) {
  int row = blockIdx.x * 8 + (threadIdx.x >> 5);
  if (row >= nrows) return;
  int lane = threadIdx.x & 31, c = lane * 4;
  f4 v = *(f4*)(X + (size_t)row * DD + c);
#pragma unroll
  for (int u = 0; u < 4; ++u) if (v[u] == 0.f) v[u] = 1e-6f;
  float ss = waveRedSum(v[0]*v[0] + v[1]*v[1] + v[2]*v[2] + v[3]*v[3]);
  float inv = rsqrtf(ss);
#pragma unroll
  for (int u = 0; u < 4; ++u) v[u] *= inv;
  *(f4*)(X + (size_t)row * DD + c) = v;
}

// cur = relu(LN((num/denom + cur)*0.5)*g + b); denom = dot(q_row, ksum) + N
__global__ __launch_bounds__(256) void attn_combine_rows(
    float* __restrict__ cur, const float* __restrict__ num,
    const float* __restrict__ q, const float* __restrict__ ksum,
    const float* __restrict__ g, const float* __restrict__ b, int nrows) {
  int row = blockIdx.x * 8 + (threadIdx.x >> 5);
  if (row >= nrows) return;
  int lane = threadIdx.x & 31, c = lane * 4;
  f4 qv = *(const f4*)(q + (size_t)row * DD + c);
  f4 ks = *(const f4*)(ksum + c);
  float denom = waveRedSum(qv[0]*ks[0] + qv[1]*ks[1] + qv[2]*ks[2] + qv[3]*ks[3])
              + (float)N_NODES;
  f4 nv = *(const f4*)(num + (size_t)row * DD + c);
  f4 cv = *(const f4*)(cur + (size_t)row * DD + c);
  f4 v;
#pragma unroll
  for (int u = 0; u < 4; ++u) v[u] = (nv[u] / denom + cv[u]) * 0.5f;
  float mu = waveRedSum(v[0] + v[1] + v[2] + v[3]) * (1.f / 128.f);
  float d0 = v[0]-mu, d1 = v[1]-mu, d2 = v[2]-mu, d3 = v[3]-mu;
  float var = waveRedSum(d0*d0 + d1*d1 + d2*d2 + d3*d3) * (1.f / 128.f);
  float rs = rsqrtf(var + 1e-5f);
  f4 o;
  o[0] = fmaxf(d0*rs*g[c+0] + b[c+0], 0.f);
  o[1] = fmaxf(d1*rs*g[c+1] + b[c+1], 0.f);
  o[2] = fmaxf(d2*rs*g[c+2] + b[c+2], 0.f);
  o[3] = fmaxf(d3*rs*g[c+3] + b[c+3], 0.f);
  *(f4*)(cur + (size_t)row * DD + c) = o;
}

// --------------------------------------------------------------- GCN pieces
__global__ void deg_edges(const int* __restrict__ col, int E, float* __restrict__ deg) {
  int i = blockIdx.x * 256 + threadIdx.x;
  if (i < E) atomicAdd(&deg[col[i]], 1.f);
}

__global__ void rsqrt_k(const float* __restrict__ deg, float* __restrict__ dinv, int n) {
  int i = blockIdx.x * 256 + threadIdx.x;
  if (i < n) dinv[i] = rsqrtf(deg[i]);
}

// one wave per (edge or self-loop); lane handles 4 feature columns
__global__ __launch_bounds__(256) void scatter_edges(
    const float* __restrict__ xw, const int* __restrict__ rows,
    const int* __restrict__ cols, const float* __restrict__ dinv,
    float* __restrict__ agg, int E, int N) {
  int t = blockIdx.x * 256 + threadIdx.x;
  int idx = t >> 5;
  if (idx >= E + N) return;
  int lane = t & 31;
  int r, c;
  if (idx < E) { r = rows[idx]; c = cols[idx]; }
  else         { r = c = idx - E; }
  float nrm = dinv[r] * dinv[c];
  int d0 = lane * 4;
  f4 v = *(const f4*)(xw + (size_t)r * DD + d0);
  atomicAdd(&agg[(size_t)c * DD + d0 + 0], nrm * v[0]);
  atomicAdd(&agg[(size_t)c * DD + d0 + 1], nrm * v[1]);
  atomicAdd(&agg[(size_t)c * DD + d0 + 2], nrm * v[2]);
  atomicAdd(&agg[(size_t)c * DD + d0 + 3], nrm * v[3]);
}

// out = relu((agg + convb[col]) * bnG[col]/sqrt(1+eps) + bnB[col]) + last
__global__ void ebn_relu_add(const float* __restrict__ agg, const float* __restrict__ convb,
                             const float* __restrict__ bg, const float* __restrict__ bb,
                             const float* __restrict__ last, float* __restrict__ out, int n) {
  int i = blockIdx.x * 256 + threadIdx.x;
  if (i >= n) return;
  int col = i & 127;
  float v = agg[i] + convb[col];
  v = fmaxf(v * (bg[col] * kBN) + bb[col], 0.f);
  out[i] = v + last[i];
}

__global__ void fuse_half(const float* __restrict__ a, const float* __restrict__ b,
                          float* __restrict__ o, int n) {
  int i = blockIdx.x * 256 + threadIdx.x;
  if (i < n) o[i] = 0.5f * (a[i] + b[i]);
}

// ---------------------------------------------------------------------------
extern "C" void kernel_launch(void* const* d_in, const int* in_sizes, int n_in,
                              void* d_out, int out_size, void* d_ws, size_t ws_size,
                              hipStream_t stream) {
  const float* x     = (const float*)d_in[0];
  const int*   ei    = (const int*)  d_in[1];
  const float* tfcw  = (const float*)d_in[2];
  const float* tfcb  = (const float*)d_in[3];
  const float* tlng  = (const float*)d_in[4];
  const float* tlnb  = (const float*)d_in[5];
  const float* qkvw  = (const float*)d_in[6];
  const float* gfcw  = (const float*)d_in[7];
  const float* gfcb  = (const float*)d_in[8];
  const float* bng   = (const float*)d_in[9];
  const float* bnb   = (const float*)d_in[10];
  const float* convw = (const float*)d_in[11];
  const float* convb = (const float*)d_in[12];
  const float* fcw   = (const float*)d_in[13];
  const float* fcb   = (const float*)d_in[14];
  const int* erow = ei;
  const int* ecol = ei + N_EDGES;

  // workspace layout (all 16B-aligned)
  const size_t SZ = (size_t)NPAD * DD;             // floats per big buffer
  float* B0   = (float*)d_ws;                      // cur / x1
  float* B1   = B0 + SZ;                           // q   / g
  float* B2   = B1 + SZ;                           // k   / last(g0)
  float* B3   = B2 + SZ;                           // v   / xw / fused
  float* B4   = B3 + SZ;                           // num / agg
  float* kv   = B4 + SZ;                           // 128*128
  float* ksum = kv + DD * DD;                      // 128
  float* deg  = ksum + DD;                         // N
  float* dinv = deg + N_NODES;                     // N
  _Float16* WT = (_Float16*)(dinv + N_NODES);      // 13 * 128*128 halves
  auto WTi = [&](int i) { return WT + (size_t)i * DD * DD; };
  _Float16* kvT = WTi(12);

  const int GB = N_NODES / 16;   // 3125 gemm blocks
  const int RB = N_NODES / 8;    // 6250 row blocks
  const int NF = N_NODES * DD;   // 6.4M elements
  const int EB = (NF + 255) / 256;

  // ---- weight prep: transpose + f16 convert ----
  wprep<<<64, 256, 0, stream>>>(tfcw, WTi(0));
  for (int i = 0; i < 6; ++i)
    wprep<<<64, 256, 0, stream>>>(qkvw + (size_t)i * DD * DD, WTi(1 + i));
  wprep<<<64, 256, 0, stream>>>(gfcw, WTi(7));
  for (int l = 0; l < 3; ++l)
    wprep<<<64, 256, 0, stream>>>(convw + (size_t)l * DD * DD, WTi(8 + l));
  wprep<<<64, 256, 0, stream>>>(fcw, WTi(11));

  // zero pad rows (N..NPAD) of k/v buffers used by kv_accum
  const int PADN = (NPAD - N_NODES) * DD;
  fill_f32<<<(PADN + 255) / 256, 256, 0, stream>>>(B2 + (size_t)N_NODES * DD, PADN, 0.f);
  fill_f32<<<(PADN + 255) / 256, 256, 0, stream>>>(B3 + (size_t)N_NODES * DD, PADN, 0.f);

  // ---- transformer branch ----
  gemm_nt128<<<GB, 256, 0, stream>>>(x, WTi(0), tfcb, nullptr, nullptr, nullptr, 0.f, B0, 1);
  ln_relu_rows<<<RB, 256, 0, stream>>>(B0, tlng, tlnb, N_NODES);
  for (int i = 0; i < 2; ++i) {
    gemm_nt128<<<GB, 256, 0, stream>>>(B0, WTi(1 + i * 3 + 0), nullptr, nullptr, nullptr, nullptr, 0.f, B1, 0);
    gemm_nt128<<<GB, 256, 0, stream>>>(B0, WTi(1 + i * 3 + 1), nullptr, nullptr, nullptr, nullptr, 0.f, B2, 0);
    gemm_nt128<<<GB, 256, 0, stream>>>(B0, WTi(1 + i * 3 + 2), nullptr, nullptr, nullptr, nullptr, 0.f, B3, 0);
    qknorm_rows<<<RB, 256, 0, stream>>>(B1, N_NODES);
    qknorm_rows<<<RB, 256, 0, stream>>>(B2, N_NODES);
    fill_f32<<<64, 256, 0, stream>>>(kv, DD * DD, 0.f);
    kv_accum<<<128, 256, 0, stream>>>(B2, B3, kv, NPAD / 64);
    ksum128<<<DD, 256, 0, stream>>>(B2, ksum, N_NODES);
    wprep<<<64, 256, 0, stream>>>(kv, kvT);
    gemm_nt128<<<GB, 256, 0, stream>>>(B1, kvT, nullptr, nullptr, nullptr, B3,
                                       (float)N_NODES, B4, 4);
    attn_combine_rows<<<RB, 256, 0, stream>>>(B0, B4, B1, ksum,
                                              tlng + (i + 1) * DD, tlnb + (i + 1) * DD, N_NODES);
  }
  // B0 == x1

  // ---- GCN branch ----
  fill_f32<<<(N_NODES + 255) / 256, 256, 0, stream>>>(deg, N_NODES, 1.f);  // self-loops
  deg_edges<<<(N_EDGES + 255) / 256, 256, 0, stream>>>(ecol, N_EDGES, deg);
  rsqrt_k<<<(N_NODES + 255) / 256, 256, 0, stream>>>(deg, dinv, N_NODES);

  gemm_nt128<<<GB, 256, 0, stream>>>(x, WTi(7), gfcb, bng, bnb, nullptr, 0.f, B1, 1 | 2);
  copy_f32<<<EB, 256, 0, stream>>>(B1, B2, NF);   // last = g0 (frozen)
  for (int l = 0; l < 3; ++l) {
    gemm_nt128<<<GB, 256, 0, stream>>>(B1, WTi(8 + l), nullptr, nullptr, nullptr, nullptr, 0.f, B3, 0);
    fill_f32<<<EB, 256, 0, stream>>>(B4, NF, 0.f);
    scatter_edges<<<((N_EDGES + N_NODES) * 32 + 255) / 256, 256, 0, stream>>>(
        B3, erow, ecol, dinv, B4, N_EDGES, N_NODES);
    ebn_relu_add<<<EB, 256, 0, stream>>>(B4, convb + l * DD, bng + (l + 1) * DD,
                                         bnb + (l + 1) * DD, B2, B1, NF);
  }
  // B1 == x2

  // ---- fuse + project ----
  fuse_half<<<EB, 256, 0, stream>>>(B1, B0, B3, NF);
  gemm_nt128<<<GB, 256, 0, stream>>>(B3, WTi(11), fcb, nullptr, nullptr, nullptr, 0.f,
                                     (float*)d_out, 1);
}
